// AGRULayer_6090263626302
// MI455X (gfx1250) — matrix operand order repaired and checked by reading, not compiled, not run on hardware
//
#include <hip/hip_runtime.h>
#include <math.h>

#define UNITS 256
#define BATCH 1024
#define SEQ   512

#define BT_STRIDE 264   // bf16 elems per row of any bf16 tile in LDS (256 + 8 pad)

typedef __attribute__((ext_vector_type(16))) __bf16 v16bf;
typedef __attribute__((ext_vector_type(8)))  float  v8f;

__device__ __forceinline__ v8f wmma_bf16(v16bf a, v16bf b, v8f c) {
  // D = A(16x32 bf16) * B(32x16 bf16) + C(16x16 f32)
  return __builtin_amdgcn_wmma_f32_16x16x32_bf16(false, a, false, b, (short)0, c,
                                                 false, false);
}

// A fragment (16x32 bf16) from a bf16 tile in LDS (rows = M, stride BT_STRIDE).
// ISA layout: lane L: row M=L%16; elems 0..7 -> K=k0..k0+7, 8..15 -> K=k0+16..k0+23,
// with k0 = kb*32 + (L/16)*8. Two aligned 16B LDS loads, no conversion.
__device__ __forceinline__ v16bf load_a_frag(const __bf16* __restrict__ src,
                                             int lane, int kb) {
  const int row = lane & 15;
  const int k0  = kb * 32 + (lane >> 4) * 8;
  const __bf16* p = src + row * BT_STRIDE + k0;
  union { v16bf v; uint4 q[2]; } u;
  u.q[0] = *(const uint4*)p;          // K = k0 .. k0+7
  u.q[1] = *(const uint4*)(p + 16);   // K = k0+16 .. k0+23
  return u.v;
}

// B fragment (32x16 bf16) from a *transposed* weight in LDS: wt[n][k].
// lane = column N=lane%16; elems 0..7 -> K=k0..k0+7, 8..15 -> K=k0+16..k0+23, k0 = kb*32 + (lane/16)*8.
__device__ __forceinline__ v16bf load_b_frag(const __bf16* __restrict__ wt,
                                             int lane, int nb, int kb) {
  const int col   = nb * 16 + (lane & 15);
  const int kbase = kb * 32 + (lane >> 4) * 8;
  const __bf16* p = wt + col * BT_STRIDE + kbase;
  union { v16bf v; uint4 q[2]; } u;
  u.q[0] = *(const uint4*)p;
  u.q[1] = *(const uint4*)(p + 16);
  return u.v;
}

// Fill LDS with bf16 transpose of a 256x256 f32 weight (row-major K x N -> wt[N][K]).
__device__ __forceinline__ void fill_wt(__bf16* __restrict__ wt, const float* __restrict__ W,
                                        int tid, int nthreads) {
  for (int idx = tid; idx < (UNITS * UNITS) / 4; idx += nthreads) {
    const int k  = idx >> 6;          // 64 float4 per K-row
    const int n4 = (idx & 63) << 2;
    const float4 w = *(const float4*)(W + k * UNITS + n4);
    wt[(n4 + 0) * BT_STRIDE + k] = (__bf16)w.x;
    wt[(n4 + 1) * BT_STRIDE + k] = (__bf16)w.y;
    wt[(n4 + 2) * BT_STRIDE + k] = (__bf16)w.z;
    wt[(n4 + 3) * BT_STRIDE + k] = (__bf16)w.w;
  }
}

// ---------------------------------------------------------------------------
// Phase 1: x_r = X @ W_ir + b_ir ; x_c = X @ W_ic + b_ic, stored TRANSPOSED as
// (T, B, U) so the scan reads a contiguous block per timestep.
// 512 threads (16 waves); both weights resident in LDS as bf16^T; activations
// staged as bf16 so A fragments are pure 16B LDS loads.
// ---------------------------------------------------------------------------
__global__ __launch_bounds__(512, 1)
void agru_xproj_kernel(const float* __restrict__ X,
                       const float* __restrict__ W_ir, const float* __restrict__ b_ir,
                       const float* __restrict__ W_ic, const float* __restrict__ b_ic,
                       float* __restrict__ xr, float* __restrict__ xc) {
  extern __shared__ char smem[];
  __bf16* wt_ir = (__bf16*)smem;
  __bf16* wt_ic = wt_ir + UNITS * BT_STRIDE;
  __bf16* atile = wt_ic + UNITS * BT_STRIDE;   // [32][BT_STRIDE] bf16

  const int tid = threadIdx.x;
  fill_wt(wt_ir, W_ir, tid, 512);
  fill_wt(wt_ic, W_ic, tid, 512);

  const int wave = tid >> 5;
  const int lane = tid & 31;
  const int mb   = wave & 1;            // which 16-row half of the 32-row chunk
  const int grp  = wave >> 1;           // 0..7
  const int osel = grp >> 2;            // 0 -> x_r, 1 -> x_c
  const int nb0  = (grp & 3) * 4;       // 4 consecutive 16-col tiles per wave
  const __bf16* wt   = osel ? wt_ic : wt_ir;
  const float*  bias = osel ? b_ic : b_ir;
  float*        dst  = osel ? xc : xr;

  const int ncol = lane & 15;
  const int lh   = lane >> 4;
  float bfrag[4];
#pragma unroll
  for (int j = 0; j < 4; ++j) bfrag[j] = bias[(nb0 + j) * 16 + ncol];

  const v8f vzero = {0.f, 0.f, 0.f, 0.f, 0.f, 0.f, 0.f, 0.f};
  const int TOTCH = (BATCH * SEQ) / 32;   // 16384 chunks of 32 rows

  for (int chunk = blockIdx.x; chunk < TOTCH; chunk += gridDim.x) {
    const long long r0 = (long long)chunk * 32;

    __syncthreads();  // previous chunk's atile reads done before overwrite
    // Stage 32x256 activations into LDS, converting f32 -> bf16 on the fly.
    for (int idx = tid; idx < 32 * 64; idx += 512) {
      const int rr = idx >> 6;
      const int c4 = (idx & 63) << 2;
      const float4 v = *(const float4*)(X + (r0 + rr) * UNITS + c4);
      __bf16 t4[4] = {(__bf16)v.x, (__bf16)v.y, (__bf16)v.z, (__bf16)v.w};
      *(uint2*)(atile + rr * BT_STRIDE + c4) = *(const uint2*)t4;
    }
    // Prefetch next chunk while we compute (global_prefetch_b8).
    {
      const long long rn = (long long)(chunk + gridDim.x) * 32;
      if (rn < (long long)BATCH * SEQ) {
        const int pr = tid >> 4;              // 0..31 rows
        const int pc = (tid & 15) << 4;       // 64B granules
        __builtin_prefetch(X + (rn + pr) * UNITS + pc, 0, 0);
      }
    }
    __syncthreads();

    v8f acc[4];
#pragma unroll
    for (int j = 0; j < 4; ++j) acc[j] = vzero;

    const __bf16* asrc = atile + mb * 16 * BT_STRIDE;
#pragma unroll
    for (int kb = 0; kb < 8; ++kb) {
      const v16bf a = load_a_frag(asrc, lane, kb);
#pragma unroll
      for (int j = 0; j < 4; ++j) {
        const v16bf b = load_b_frag(wt, lane, nb0 + j, kb);
        acc[j] = wmma_bf16(a, b, acc[j]);
      }
    }

    // Epilogue: bias add + non-temporal store, transposed to (t, b, n).
    const int b_idx = (int)(r0 >> 9);       // r0 / SEQ  (32 | 512 -> single b per chunk)
    const int t0    = (int)(r0 & 511);      // r0 % SEQ
#pragma unroll
    for (int j = 0; j < 4; ++j) {
      const int n = (nb0 + j) * 16 + ncol;
#pragma unroll
      for (int r = 0; r < 8; ++r) {
        const int trow = t0 + mb * 16 + lh * 8 + r;
        __builtin_nontemporal_store(acc[j][r] + bfrag[j],
            dst + ((size_t)trow * BATCH + b_idx) * UNITS + n);
      }
    }
  }
}

// ---------------------------------------------------------------------------
// Phase 2: sequential AGRU scan. 64 WGs x 512 threads; wave w owns N-tile w.
// W_hr^T, W_hc^T resident in LDS (bf16). h carried in f32 registers (fragment
// layout); bf16 shadow of h / reset*h in LDS feeds the WMMA A fragments.
// ---------------------------------------------------------------------------
__global__ __launch_bounds__(512, 1)
void agru_scan_kernel(const float* __restrict__ xr, const float* __restrict__ xc,
                      const float* __restrict__ att,
                      const float* __restrict__ W_hr, const float* __restrict__ b_hr,
                      const float* __restrict__ W_hc, const float* __restrict__ b_hc,
                      float* __restrict__ out) {
  extern __shared__ char smem[];
  __bf16* wt_hr = (__bf16*)smem;
  __bf16* wt_hc = wt_hr + UNITS * BT_STRIDE;
  __bf16* htile = wt_hc + UNITS * BT_STRIDE;   // [16][BT_STRIDE] bf16 shadow of h
  __bf16* rhtil = htile + 16 * BT_STRIDE;      // [16][BT_STRIDE] bf16 reset*h

  const int tid = threadIdx.x;
  fill_wt(wt_hr, W_hr, tid, 512);
  fill_wt(wt_hc, W_hc, tid, 512);
  for (int i = tid; i < 16 * BT_STRIDE; i += 512) htile[i] = (__bf16)0.f;
  __syncthreads();

  const int wave = tid >> 5;     // n-block 0..15
  const int lane = tid & 31;
  const int nb   = wave;
  const int ncol = lane & 15;
  const int lh   = lane >> 4;
  const int n    = nb * 16 + ncol;
  const float bhr = b_hr[n];
  const float bhc = b_hc[n];
  const int r0 = blockIdx.x * 16;
  const v8f vzero = {0.f, 0.f, 0.f, 0.f, 0.f, 0.f, 0.f, 0.f};

  // Full-precision h for this lane's fragment elements (M = lh*8+r, N = n).
  float hreg[8];
#pragma unroll
  for (int r = 0; r < 8; ++r) hreg[r] = 0.f;

  for (int t = 0; t < SEQ; ++t) {
    const size_t tbase = (size_t)t * BATCH * UNITS;

    // Prefetch next step's contiguous x_r/x_c block (one 64B granule/thread).
    if (t + 1 < SEQ) {
      const float* pb = (tid >> 8) ? xc : xr;
      const int prow  = (tid >> 4) & 15;
      const int pcol  = (tid & 15) << 4;
      __builtin_prefetch(pb + tbase + (size_t)BATCH * UNITS
                            + (size_t)(r0 + prow) * UNITS + pcol, 0, 0);
    }

    // Streamed per-step inputs in C/D fragment layout.
    float xrf[8], xcf[8], af[8];
#pragma unroll
    for (int r = 0; r < 8; ++r) {
      const int m = lh * 8 + r;
      const size_t gx = tbase + (size_t)(r0 + m) * UNITS + n;
      xrf[r] = __builtin_nontemporal_load(xr + gx);
      xcf[r] = __builtin_nontemporal_load(xc + gx);
      af[r]  = att[(size_t)(r0 + m) * SEQ + t];
    }

    // GEMM1: h @ W_hr  (two independent accumulation chains)
    v8f acc0 = vzero, acc1 = vzero;
#pragma unroll
    for (int kb = 0; kb < 8; kb += 2) {
      acc0 = wmma_bf16(load_a_frag(htile, lane, kb),
                       load_b_frag(wt_hr, lane, nb, kb), acc0);
      acc1 = wmma_bf16(load_a_frag(htile, lane, kb + 1),
                       load_b_frag(wt_hr, lane, nb, kb + 1), acc1);
    }
    const v8f acc = acc0 + acc1;

    // reset gate; write bf16 reset*h for GEMM2 (each wave owns its 16 columns)
#pragma unroll
    for (int r = 0; r < 8; ++r) {
      const int m = lh * 8 + r;
      const float z = xrf[r] + acc[r] + bhr;
      const float reset = 1.f / (1.f + __expf(-z));
      rhtil[m * BT_STRIDE + n] = (__bf16)(reset * hreg[r]);
    }
    __syncthreads();   // rh ready; all htile reads of this step complete

    // GEMM2: (reset*h) @ W_hc
    v8f acd0 = vzero, acd1 = vzero;
#pragma unroll
    for (int kb = 0; kb < 8; kb += 2) {
      acd0 = wmma_bf16(load_a_frag(rhtil, lane, kb),
                       load_b_frag(wt_hc, lane, nb, kb), acd0);
      acd1 = wmma_bf16(load_a_frag(rhtil, lane, kb + 1),
                       load_b_frag(wt_hc, lane, nb, kb + 1), acd1);
    }
    const v8f acd = acd0 + acd1;

    // candidate + attention-gated update of h (f32 in registers, bf16 shadow)
#pragma unroll
    for (int r = 0; r < 8; ++r) {
      const int m = lh * 8 + r;
      const float cand = tanhf(xcf[r] + acd[r] + bhc);
      const float a_t  = af[r];
      hreg[r] = (1.f - a_t) * hreg[r] + a_t * cand;
      htile[m * BT_STRIDE + n] = (__bf16)hreg[r];
    }
    __syncthreads();   // h shadow ready for next step
  }

  // Final hidden state straight from registers (coalesced across lanes per r).
#pragma unroll
  for (int r = 0; r < 8; ++r) {
    const int m = lh * 8 + r;
    out[(size_t)(r0 + m) * UNITS + n] = hreg[r];
  }
}

extern "C" void kernel_launch(void* const* d_in, const int* in_sizes, int n_in,
                              void* d_out, int out_size, void* d_ws, size_t ws_size,
                              hipStream_t stream) {
  const float* X    = (const float*)d_in[0];  // interest_states (B,T,U)
  const float* att  = (const float*)d_in[1];  // attention_scores (B,T,1)
  const float* W_ir = (const float*)d_in[2];
  const float* b_ir = (const float*)d_in[3];
  const float* W_hr = (const float*)d_in[4];
  const float* b_hr = (const float*)d_in[5];
  const float* W_ic = (const float*)d_in[6];
  const float* b_ic = (const float*)d_in[7];
  const float* W_hc = (const float*)d_in[8];
  const float* b_hc = (const float*)d_in[9];
  float* out = (float*)d_out;

  float* xr = (float*)d_ws;                          // (T, B, U) f32
  float* xc = xr + (size_t)BATCH * SEQ * UNITS;      // (T, B, U) f32

  const size_t wbytes = (size_t)2 * UNITS * BT_STRIDE * sizeof(__bf16); // 270336
  const size_t smem1  = wbytes + (size_t)32 * BT_STRIDE * sizeof(__bf16);
  const size_t smem2  = wbytes + (size_t)2 * 16 * BT_STRIDE * sizeof(__bf16);

  // Opt in to >64KB dynamic LDS (CDNA5 WGP has 320KB).
  hipFuncSetAttribute((const void*)agru_xproj_kernel,
                      hipFuncAttributeMaxDynamicSharedMemorySize, (int)smem1);
  hipFuncSetAttribute((const void*)agru_scan_kernel,
                      hipFuncAttributeMaxDynamicSharedMemorySize, (int)smem2);

  agru_xproj_kernel<<<2048, 512, smem1, stream>>>(X, W_ir, b_ir, W_ic, b_ic, xr, xc);
  agru_scan_kernel<<<BATCH / 16, 512, smem2, stream>>>(xr, xc, att,
                                                       W_hr, b_hr, W_hc, b_hc, out);
}
